// SelfAttentionForMoe_27779848471414
// MI455X (gfx1250) — matrix operand order repaired and checked by reading, not compiled, not run on hardware
//
#include <hip/hip_runtime.h>

typedef __attribute__((ext_vector_type(16))) _Float16 v16h;
typedef __attribute__((ext_vector_type(4)))  _Float16 v4h;
typedef __attribute__((ext_vector_type(8)))  float    v8f;
typedef __attribute__((ext_vector_type(4)))  int      v4i;

// ---------------------------------------------------------------------------
// CDNA5 async global->LDS copy path (guarded; falls back to float4 copy).
// Builtin signature (verified by hipcc): (v4i AS1*, v4i AS3*, Ii, Ii)
// ---------------------------------------------------------------------------
#if defined(__HIP_DEVICE_COMPILE__) && __has_builtin(__builtin_amdgcn_global_load_async_to_lds_b128)
#define USE_ASYNC_LDS 1
#endif

__device__ __forceinline__ void cp16_g2l(void* lds, const void* g) {
#ifdef USE_ASYNC_LDS
  auto gp = (__attribute__((address_space(1))) v4i*)(unsigned long long)g;
  auto lp = (__attribute__((address_space(3))) v4i*)(unsigned int)(unsigned long long)lds;
  __builtin_amdgcn_global_load_async_to_lds_b128(gp, lp, 0, 0);
#else
  *(float4*)lds = *(const float4*)g;
#endif
}

__device__ __forceinline__ void cp_wait_async() {
#ifdef USE_ASYNC_LDS
#if __has_builtin(__builtin_amdgcn_s_wait_asynccnt)
  __builtin_amdgcn_s_wait_asynccnt(0);
#else
  asm volatile("s_wait_asynccnt 0" ::: "memory");
#endif
#endif
}

// ---------------------------------------------------------------------------
// WMMA helpers (CDNA5 wave32, 16x16x32 f16 -> f32)
// ---------------------------------------------------------------------------
__device__ __forceinline__ v8f wmma16x16x32(v16h a, v16h b, v8f c) {
  return __builtin_amdgcn_wmma_f32_16x16x32_f16(
      false, a, false, b, (short)0, c, false, false);
}

__device__ __forceinline__ v8f zero8() {
  v8f z;
#pragma unroll
  for (int i = 0; i < 8; ++i) z[i] = 0.0f;
  return z;
}

// A fragment: 16(M) x 32(K) f16, element (m,k) at base[m*ld + k]
__device__ __forceinline__ v16h frag_a_rowmajor(const _Float16* base, int ld) {
  const int lane = threadIdx.x & 31;
  const _Float16* row = base + (lane & 15) * ld;
  const int kh = (lane >> 4) << 3;  // 0 or 8
  v16h a;
#pragma unroll
  for (int v = 0; v < 8; ++v) {
    int k = ((v & 3) << 1) + kh + ((v >> 2) << 4);
    a[2 * v]     = row[k];
    a[2 * v + 1] = row[k + 1];
  }
  return a;
}

// B fragment: 32(K) x 16(N), element (k,n) at base[n*ld + k]  (K contiguous)
__device__ __forceinline__ v16h frag_b_kcontig(const _Float16* base, int ld) {
  const int lane = threadIdx.x & 31;
  const _Float16* col = base + (lane & 15) * ld;
  const int kh = (lane >> 4) << 3;  // 0 or 8
  v16h b;
#pragma unroll
  for (int v = 0; v < 8; ++v) {
    int k = ((v & 3) << 1) + kh + ((v >> 2) << 4);
    b[2 * v]     = col[k];
    b[2 * v + 1] = col[k + 1];
  }
  return b;
}

// ---------------------------------------------------------------------------
// Tiled WMMA GEMM, software-pipelined: global loads for tile k+1 overlap the
// WMMA compute of tile k. Block tile 128x128, 8 waves, wave tile 32x64.
// EPI==0: QKV epilogue (adds b1, scatters q (pre-scaled), k, and V TRANSPOSED)
// EPI==1: out epilogue (adds b2, fp32 store)
// ---------------------------------------------------------------------------
template <typename AT, int EPI>
__global__ __launch_bounds__(256)
void gemm128(const AT* __restrict__ A, const float* __restrict__ Bm,
             const float* __restrict__ bias, int M, int N, int K,
             _Float16* __restrict__ qb, _Float16* __restrict__ kb,
             _Float16* __restrict__ vb, float* __restrict__ outp) {
  constexpr int LDA = 40;  // f16 elements; 80B rows (16B-aligned)
  __shared__ __align__(16) _Float16 Ah[128 * LDA];
  __shared__ __align__(16) _Float16 Bt[128 * LDA];
  const int tid  = threadIdx.x;
  const int lane = tid & 31, wid = tid >> 5;
  const int m0 = blockIdx.y * 128, n0 = blockIdx.x * 128;
  const int wm = (wid & 3) * 32, wn = (wid >> 2) * 64;

  v8f acc[2][4];
#pragma unroll
  for (int i = 0; i < 2; ++i)
#pragma unroll
    for (int j = 0; j < 4; ++j) acc[i][j] = zero8();

  float4 aR[4], bR[4];
  auto load_tile = [&](int k0) {
    if (sizeof(AT) == 4) {
#pragma unroll
      for (int i = 0; i < 4; ++i) {
        int c = tid + (i << 8), r = c >> 3, c4 = (c & 7) << 2;
        aR[i] = *(const float4*)(const void*)&A[(size_t)(m0 + r) * K + k0 + c4];
      }
    } else {
#pragma unroll
      for (int i = 0; i < 2; ++i) {
        int c = tid + (i << 8), r = c >> 2, c8 = (c & 3) << 3;
        aR[i] = *(const float4*)(const void*)&A[(size_t)(m0 + r) * K + k0 + c8];
      }
    }
#pragma unroll
    for (int i = 0; i < 4; ++i) {
      int c = tid + (i << 8), kr = c >> 5, nc = (c & 31) << 2;
      bR[i] = *(const float4*)&Bm[(size_t)(k0 + kr) * N + n0 + nc];
    }
  };
  auto store_tile = [&]() {
    if (sizeof(AT) == 4) {
#pragma unroll
      for (int i = 0; i < 4; ++i) {
        int c = tid + (i << 8), r = c >> 3, c4 = (c & 7) << 2;
        v4h h = {(_Float16)aR[i].x, (_Float16)aR[i].y, (_Float16)aR[i].z, (_Float16)aR[i].w};
        *(v4h*)&Ah[r * LDA + c4] = h;
      }
    } else {
#pragma unroll
      for (int i = 0; i < 2; ++i) {
        int c = tid + (i << 8), r = c >> 2, c8 = (c & 3) << 3;
        *(float4*)&Ah[r * LDA + c8] = aR[i];
      }
    }
#pragma unroll
    for (int i = 0; i < 4; ++i) {
      int c = tid + (i << 8), kr = c >> 5, nc = (c & 31) << 2;
      Bt[(nc + 0) * LDA + kr] = (_Float16)bR[i].x;
      Bt[(nc + 1) * LDA + kr] = (_Float16)bR[i].y;
      Bt[(nc + 2) * LDA + kr] = (_Float16)bR[i].z;
      Bt[(nc + 3) * LDA + kr] = (_Float16)bR[i].w;
    }
  };

  const int ksteps = K >> 5;
  load_tile(0);
  for (int kk = 0; kk < ksteps; ++kk) {
    store_tile();
    __syncthreads();
    if (kk + 1 < ksteps) load_tile((kk + 1) << 5);  // overlap next loads w/ compute
    if (kk + 2 < ksteps) {                          // global_prefetch_b8, distance 2
      int kp = (kk + 2) << 5;
      __builtin_prefetch(&A[(size_t)(m0 + (tid >> 1)) * K + kp], 0, 0);
      __builtin_prefetch(&Bm[(size_t)(kp + (tid >> 3)) * N + n0 + ((tid & 7) << 4)], 0, 0);
    }
    v16h af[2], bf[4];
#pragma unroll
    for (int mt = 0; mt < 2; ++mt) af[mt] = frag_a_rowmajor(&Ah[(wm + mt * 16) * LDA], LDA);
#pragma unroll
    for (int nt = 0; nt < 4; ++nt) bf[nt] = frag_b_kcontig(&Bt[(wn + nt * 16) * LDA], LDA);
#pragma unroll
    for (int mt = 0; mt < 2; ++mt)
#pragma unroll
      for (int nt = 0; nt < 4; ++nt)
        acc[mt][nt] = wmma16x16x32(af[mt], bf[nt], acc[mt][nt]);
    __syncthreads();
  }

  // Epilogue. C layout: VGPR r -> row r (lanes 0-15) or r+8 (lanes 16-31).
  const int nl = lane & 15, ro = (lane >> 4) << 3;
#pragma unroll
  for (int mt = 0; mt < 2; ++mt)
#pragma unroll
    for (int nt = 0; nt < 4; ++nt)
#pragma unroll
      for (int r = 0; r < 8; ++r) {
        const int mrow = m0 + wm + mt * 16 + r + ro;
        const int ncol = n0 + wn + nt * 16 + nl;
        const float val = acc[mt][nt][r] + bias[ncol];
        if (EPI == 0) {
          const int t = ncol >> 10;
          const int rem = ncol & 1023;
          const int hh = rem >> 6, dd = rem & 63;
          const int bb = mrow >> 11, ss = mrow & 2047;
          if (t == 0) {        // q: [B,NH,S,DK], pre-scaled by 1/sqrt(64)
            qb[(((size_t)bb * 16 + hh) * 2048 + ss) * 64 + dd] = (_Float16)(val * 0.125f);
          } else if (t == 1) { // k: [B,NH,S,DK]
            kb[(((size_t)bb * 16 + hh) * 2048 + ss) * 64 + dd] = (_Float16)val;
          } else {             // v TRANSPOSED: [B,NH,DK,S]
            vb[(((size_t)bb * 16 + hh) * 64 + dd) * 2048 + ss] = (_Float16)val;
          }
        } else {
          outp[(size_t)mrow * N + ncol] = val;
        }
      }
}

// ---------------------------------------------------------------------------
// Flash attention, double-buffered async staging. Wave = 16 query rows,
// 4 waves/block share (b,h). 64-key blocks; K [key][d], V^T [d][key].
// Row-sum l computed by WMMA against an all-ones B fragment (5th C tile).
// ---------------------------------------------------------------------------
__global__ __launch_bounds__(128)
void attn_flash(const _Float16* __restrict__ qg, const _Float16* __restrict__ kg,
                const _Float16* __restrict__ vtg, _Float16* __restrict__ ctx) {
  constexpr int LDK = 72;  // padded stride: 144B rows (16B-aligned, bank-skewed)
  __shared__ __align__(16) _Float16 Kt[2][64 * LDK];
  __shared__ __align__(16) _Float16 Vt[2][64 * LDK];
  __shared__ __align__(16) _Float16 Pst[4][16 * LDK];

  const int tid = threadIdx.x, lane = tid & 31, wid = tid >> 5;
  const int b = blockIdx.z, h = blockIdx.y;
  const int qstart = blockIdx.x * 64 + wid * 16;
  const size_t bh = (size_t)(b * 16 + h);

  const _Float16* qbase = qg + (bh * 2048 + (size_t)qstart) * 64;
  const v16h qf0 = frag_a_rowmajor(qbase, 64);       // d = 0..31
  const v16h qf1 = frag_a_rowmajor(qbase + 32, 64);  // d = 32..63

  v16h ones;
#pragma unroll
  for (int i = 0; i < 16; ++i) ones[i] = (_Float16)1.0f;

  v8f accv[5];  // [0..3] ctx d-tiles, [4] running row-sum l
#pragma unroll
  for (int nt = 0; nt < 5; ++nt) accv[nt] = zero8();
  float mrun[8];
#pragma unroll
  for (int r = 0; r < 8; ++r) mrun[r] = -1e30f;

  const _Float16* kbh  = kg  + bh * 2048 * 64;   // [s][d]
  const _Float16* vtbh = vtg + bh * 64 * 2048;   // [d][s]

  auto stage = [&](int kb0, int buf) {
    _Float16* Kd = Kt[buf];
    _Float16* Vd = Vt[buf];
#pragma unroll
    for (int i = 0; i < 4; ++i) {
      int c = tid + (i << 7), row = c >> 3, col8 = (c & 7) << 3;
      cp16_g2l(&Kd[row * LDK + col8], &kbh[(size_t)(kb0 + row) * 64 + col8]);
      cp16_g2l(&Vd[row * LDK + col8], &vtbh[(size_t)row * 2048 + kb0 + col8]);
    }
  };

  stage(0, 0);
  for (int it = 0; it < 32; ++it) {            // 32 blocks of 64 keys
    cp_wait_async();
    __syncthreads();                           // buf[it&1] ready for all waves
    if (it + 1 < 32) stage((it + 1) * 64, (it + 1) & 1);  // DMA overlaps compute
    const _Float16* Kc = Kt[it & 1];
    const _Float16* Vc = Vt[it & 1];

    // scores S[16x64]: 4 key-tiles x 2 d-chunks
    v8f s[4];
#pragma unroll
    for (int nt = 0; nt < 4; ++nt) {
      const _Float16* kb_ = &Kc[(nt * 16) * LDK];
      s[nt] = wmma16x16x32(qf0, frag_b_kcontig(kb_, LDK), zero8());
      s[nt] = wmma16x16x32(qf1, frag_b_kcontig(kb_ + 32, LDK), s[nt]);
    }

    // batched row-max: 4 xor steps of 8 independent shuffles each
    float t8[8];
#pragma unroll
    for (int r = 0; r < 8; ++r)
      t8[r] = fmaxf(fmaxf(s[0][r], s[1][r]), fmaxf(s[2][r], s[3][r]));
#pragma unroll
    for (int st = 1; st <= 8; st <<= 1) {
      float u[8];
#pragma unroll
      for (int r = 0; r < 8; ++r) u[r] = __shfl_xor(t8[r], st, 32);
#pragma unroll
      for (int r = 0; r < 8; ++r) t8[r] = fmaxf(t8[r], u[r]);
    }

    float alpha[8];
#pragma unroll
    for (int r = 0; r < 8; ++r) {
      const float mn = fmaxf(mrun[r], t8[r]);
      alpha[r] = __expf(mrun[r] - mn);
      mrun[r] = mn;
#pragma unroll
      for (int nt = 0; nt < 4; ++nt) s[nt][r] = __expf(s[nt][r] - mn);
    }

    // P: C-layout -> per-wave LDS -> A-fragments (in-order DS within wave)
    _Float16* pw = Pst[wid];
    {
      const int nl = lane & 15, ro = (lane >> 4) << 3;
#pragma unroll
      for (int nt = 0; nt < 4; ++nt)
#pragma unroll
        for (int r = 0; r < 8; ++r)
          pw[(r + ro) * LDK + nt * 16 + nl] = (_Float16)s[nt][r];
    }
#pragma unroll
    for (int dt = 0; dt < 5; ++dt)
#pragma unroll
      for (int r = 0; r < 8; ++r) accv[dt][r] *= alpha[r];

    const v16h pf0 = frag_a_rowmajor(pw, LDK);       // keys 0..31
    const v16h pf1 = frag_a_rowmajor(pw + 32, LDK);  // keys 32..63
#pragma unroll
    for (int dt = 0; dt < 4; ++dt) {
      const _Float16* vb_ = &Vc[(dt * 16) * LDK];
      accv[dt] = wmma16x16x32(pf0, frag_b_kcontig(vb_, LDK), accv[dt]);
      accv[dt] = wmma16x16x32(pf1, frag_b_kcontig(vb_ + 32, LDK), accv[dt]);
    }
    // running row-sum via WMMA against all-ones B (no cross-lane reduction)
    accv[4] = wmma16x16x32(pf0, ones, accv[4]);
    accv[4] = wmma16x16x32(pf1, ones, accv[4]);
  }

  // normalize and store ctx f16 [B, S, NH*DK]
  const int nl = lane & 15, ro = (lane >> 4) << 3;
  float inv[8];
#pragma unroll
  for (int r = 0; r < 8; ++r) inv[r] = 1.0f / accv[4][r];
#pragma unroll
  for (int dt = 0; dt < 4; ++dt)
#pragma unroll
    for (int r = 0; r < 8; ++r) {
      const int row = qstart + r + ro;
      const int col = h * 64 + dt * 16 + nl;
      ctx[((size_t)b * 2048 + row) * 1024 + col] = (_Float16)(accv[dt][r] * inv[r]);
    }
}

// ---------------------------------------------------------------------------
// Launch: GEMM1 (QKV, v transposed) -> flash attention -> GEMM2 (out proj)
// ---------------------------------------------------------------------------
extern "C" void kernel_launch(void* const* d_in, const int* in_sizes, int n_in,
                              void* d_out, int out_size, void* d_ws, size_t ws_size,
                              hipStream_t stream) {
  (void)in_sizes; (void)n_in; (void)out_size; (void)ws_size;
  const float* x  = (const float*)d_in[0];   // [4,2048,1024]
  const float* W1 = (const float*)d_in[1];   // [1024,3072]
  const float* b1 = (const float*)d_in[2];   // [3072]
  const float* W2 = (const float*)d_in[3];   // [1024,1024]
  const float* b2 = (const float*)d_in[4];   // [1024]
  float* out = (float*)d_out;                // [4,2048,1024]

  const size_t nqkv = (size_t)4 * 16 * 2048 * 64;  // 8M elements per buffer
  _Float16* qh = (_Float16*)d_ws;
  _Float16* kh = qh + nqkv;
  _Float16* vh = kh + nqkv;      // stored transposed [B,NH,DK,S]
  _Float16* ch = vh + nqkv;      // ctx [8192,1024] f16

  gemm128<float, 0><<<dim3(3072 / 128, 8192 / 128), 256, 0, stream>>>(
      x, W1, b1, 8192, 3072, 1024, qh, kh, vh, nullptr);

  attn_flash<<<dim3(32, 16, 4), 128, 0, stream>>>(qh, kh, vh, ch);

  gemm128<_Float16, 1><<<dim3(1024 / 128, 8192 / 128), 256, 0, stream>>>(
      ch, W2, b2, 8192, 1024, 1024, nullptr, nullptr, nullptr, out);
}
